// MultiHeadAttention_28570122453511
// MI455X (gfx1250) — matrix operand order repaired and checked by reading, not compiled, not run on hardware
//
#include <hip/hip_runtime.h>
#include <hip/hip_bf16.h>
#include <stddef.h>
#include <stdint.h>

#define S_LEN   2048
#define DM      768
#define NH      12
#define DH      64
#define BATCH   2
#define ROWS    (BATCH * S_LEN)      // 4096 merged rows
#define NEG_INF (-1000000000.0f)

typedef __bf16 bf16;
typedef bf16  v16bf __attribute__((ext_vector_type(16)));
typedef bf16  v8bf  __attribute__((ext_vector_type(8)));
typedef float v8f   __attribute__((ext_vector_type(8)));
typedef float v4f   __attribute__((ext_vector_type(4)));
typedef int   v4i   __attribute__((ext_vector_type(4)));

// ---------------------------------------------------------------------------
// D = A(16x32 bf16) x B(32x16 bf16) + C(16x16 f32)   — one CDNA5 WMMA
// ---------------------------------------------------------------------------
static __device__ inline v8f wmma_bf16(v16bf a, v16bf b, v8f c) {
  return __builtin_amdgcn_wmma_f32_16x16x32_bf16(
      /*neg_a=*/false, a, /*neg_b=*/false, b,
      /*c_mod=*/(short)0, c, /*reuse_a=*/false, /*reuse_b=*/false);
}

// A-operand (16x32, 16-bit) per ISA 7.12.2:
//   lane<16 : row = lane,    K = {0..7, 16..23}
//   lane>=16: row = lane-16, K = {8..15, 24..31}
static __device__ inline v16bf load_a_bf16(const bf16* p, int ld, int row0,
                                           int k0, int lane) {
  int ll = lane & 15, lh = lane >> 4;
  const bf16* r = p + (size_t)(row0 + ll) * ld + k0 + lh * 8;
  union { v16bf v; v8bf h[2]; } u;
  u.h[0] = *(const v8bf*)(r);
  u.h[1] = *(const v8bf*)(r + 16);
  return u.v;
}

// Same A layout, converting fp32 -> bf16 on the fly.
static __device__ inline v16bf load_a_f32(const float* p, int ld, int row0,
                                          int k0, int lane) {
  int ll = lane & 15, lh = lane >> 4;
  const float* r = p + (size_t)(row0 + ll) * ld + k0 + lh * 8;
  v16bf out;
#pragma unroll
  for (int i = 0; i < 8; ++i) out[i]     = (bf16)r[i];
#pragma unroll
  for (int i = 0; i < 8; ++i) out[8 + i] = (bf16)r[16 + i];
  return out;
}

// B-operand (32x16, 16-bit) per ISA 7.12.4:
//   lane<16 : col N = lane,    K = {0..7, 16..23}
//   lane>=16: col N = lane-16, K = {8..15, 24..31}
// element(K=k, N=n) = src[(n0+n)*ld + k0 + k]  (two 8-half runs/lane)
static __device__ inline v16bf load_b_bf16(const bf16* p, int ld, int n0,
                                           int k0, int lane) {
  int ll = lane & 15, lh = lane >> 4;
  const bf16* r = p + (size_t)(n0 + ll) * ld + k0 + lh * 8;
  union { v16bf v; v8bf h[2]; } u;
  u.h[0] = *(const v8bf*)(r);
  u.h[1] = *(const v8bf*)(r + 16);
  return u.v;
}

static __device__ inline v16bf load_b_f32(const float* p, int ld, int n0,
                                          int k0, int lane) {
  int ll = lane & 15, lh = lane >> 4;
  const float* r = p + (size_t)(n0 + ll) * ld + k0 + lh * 8;
  v16bf out;
#pragma unroll
  for (int i = 0; i < 8; ++i) out[i]     = (bf16)r[i];
#pragma unroll
  for (int i = 0; i < 8; ++i) out[8 + i] = (bf16)r[16 + i];
  return out;
}

// ---------------------------------------------------------------------------
// Async global->LDS staging (CDNA5 GLOBAL_LOAD_ASYNC_TO_LDS_B128, ASYNCcnt),
// with a synchronous ds-staging fallback if the builtin is absent.
// Builtin signature (from diagnostic): (global int4* src, local int4* dst,
//                                       imm int offset, imm int cpol)
// ---------------------------------------------------------------------------
#if __has_builtin(__builtin_amdgcn_global_load_async_to_lds_b128)
#define HAVE_ASYNC_LDS 1
#else
#define HAVE_ASYNC_LDS 0
#endif

typedef __attribute__((address_space(1))) v4i gv4i;
typedef __attribute__((address_space(3))) v4i lv4i;

static __device__ inline void async_copy16(const bf16* gsrc, bf16* lds_dst) {
#if HAVE_ASYNC_LDS
  __builtin_amdgcn_global_load_async_to_lds_b128(
      (gv4i*)(uintptr_t)gsrc, (lv4i*)lds_dst, 0, 0);
#else
  *(v8bf*)lds_dst = *(const v8bf*)gsrc;     // load + ds_store fallback
#endif
}

static __device__ inline void wait_async_copies() {
#if HAVE_ASYNC_LDS
#if __has_builtin(__builtin_amdgcn_s_wait_asynccnt)
  __builtin_amdgcn_s_wait_asynccnt(0);
#else
  asm volatile("s_wait_asynccnt 0" ::: "memory");
#endif
#endif
}

// Stage one 32-wide K chunk (32x64 bf16, contiguous) and V chunk (64x32 bf16
// gathered from transposed V) into LDS; 4 async b128 per thread, block-wide.
static __device__ inline void stage_kv(const bf16* kk, const bf16* vT, int c0,
                                       bf16* ksm, bf16* vsm, int tid) {
#pragma unroll
  for (int i = 0; i < 2; ++i) {                 // K: 2048 halves contiguous
    int e = (tid + i * 128) * 8;
    async_copy16(kk + (size_t)c0 * DH + e, ksm + e);
  }
#pragma unroll
  for (int i = 0; i < 2; ++i) {                 // V: 64 rows x 32 halves
    int e = (tid + i * 128) * 8;
    int d = e >> 5, s = e & 31;
    async_copy16(vT + (size_t)d * S_LEN + c0 + s, vsm + e);
  }
}

// ---------------------------------------------------------------------------
// Kernel 1: QKV projections.  y = X @ W^T + b
// blockIdx.y: 0=Q (pre-scaled 1/8), 1=K, 2=V (stored transposed [b,h,d,S]).
// ---------------------------------------------------------------------------
__global__ __launch_bounds__(256) void proj_qkv(
    const float* __restrict__ Xq, const float* __restrict__ Xk,
    const float* __restrict__ Xv,
    const float* __restrict__ Wq, const float* __restrict__ bq,
    const float* __restrict__ Wk, const float* __restrict__ bk,
    const float* __restrict__ Wv, const float* __restrict__ bv,
    bf16* __restrict__ q_ws, bf16* __restrict__ k_ws,
    bf16* __restrict__ vT_ws) {
  const int which = blockIdx.y;
  const float* X = (which == 0) ? Xq : (which == 1) ? Xk : Xv;
  const float* W = (which == 0) ? Wq : (which == 1) ? Wk : Wv;
  const float* bias = (which == 0) ? bq : (which == 1) ? bk : bv;

  const int lane = threadIdx.x & 31;
  const int wave = threadIdx.x >> 5;
  const int tile = blockIdx.x * 8 + wave;
  const int tn = tile % (DM / 16);
  const int tm = tile / (DM / 16);
  const int r0 = tm * 16, n0 = tn * 16;
  const int ll = lane & 15, lh = lane >> 4;

  v8f acc = {};
  for (int k0 = 0; k0 < DM; k0 += 32) {
    if (k0 + 32 < DM) {   // prefetch next tiles (global_prefetch_b8)
      __builtin_prefetch(X + (size_t)(r0 + ll) * DM + k0 + 32 + lh * 8, 0, 0);
      __builtin_prefetch(W + (size_t)(n0 + ll) * DM + k0 + 32 + lh * 16, 0, 0);
    }
    v16bf a = load_a_f32(X, DM, r0, k0, lane);
    v16bf b = load_b_f32(W, DM, n0, k0, lane);   // W row-major == B layout
    acc = wmma_bf16(a, b, acc);
  }

  const int n = n0 + ll;
  const float bs = bias[n];
  const int h = n / DH, d = n % DH;

  if (which < 2) {
    bf16* dst = (which == 0) ? q_ws : k_ws;
    const float sc = (which == 0) ? 0.125f : 1.0f;   // fold 1/sqrt(d) into Q
#pragma unroll
    for (int j = 0; j < 8; ++j) {
      int row = r0 + j + lh * 8;
      int bb = row >> 11, s = row & (S_LEN - 1);
      dst[((size_t)((bb * NH + h) * S_LEN) + s) * DH + d] =
          (bf16)((acc[j] + bs) * sc);
    }
  } else {
    int row0 = r0 + lh * 8;
    int bb = row0 >> 11, s0 = row0 & (S_LEN - 1);
    v8bf pk;
#pragma unroll
    for (int j = 0; j < 8; ++j) pk[j] = (bf16)(acc[j] + bs);
    *(v8bf*)(vT_ws + ((size_t)((bb * NH + h) * DH + d) * S_LEN + s0)) = pk;
  }
}

// ---------------------------------------------------------------------------
// Kernel 2: fused scores + causal mask + online softmax + P@V.
// 4 waves/block = 4 consecutive q-tiles sharing async-staged K/V chunks
// (double buffered). Scores (masked, fp32) written to d_out exactly once.
// ---------------------------------------------------------------------------
__global__ __launch_bounds__(128) void attn_kernel(
    const bf16* __restrict__ q_ws, const bf16* __restrict__ k_ws,
    const bf16* __restrict__ vT_ws, float* __restrict__ scores,
    bf16* __restrict__ attn_ws) {
  __shared__ bf16 k_sm[2][32 * DH];   // [buf][krow 32][d 64]   4KB each
  __shared__ bf16 v_sm[2][DH * 32];   // [buf][d 64][s 32]      4KB each
  __shared__ bf16 p_lds[4][16 * 32];  // per-wave P transpose   1KB each

  const int tid  = threadIdx.x;
  const int lane = tid & 31;
  const int wave = tid >> 5;
  const int h = blockIdx.y, b = blockIdx.z;
  const int q0 = (blockIdx.x * 4 + wave) * 16;
  const int q0max = (blockIdx.x * 4 + 3) * 16 + 15;   // block causal bound
  const int ll = lane & 15, lh = lane >> 4;

  const bf16* q  = q_ws  + (size_t)((b * NH + h) * S_LEN) * DH;
  const bf16* kk = k_ws  + (size_t)((b * NH + h) * S_LEN) * DH;
  const bf16* vT = vT_ws + (size_t)((b * NH + h) * DH) * S_LEN;
  float* sc = scores + (size_t)(b * NH + h) * S_LEN * S_LEN;
  bf16* pl = p_lds[wave];

  const v16bf a_lo = load_a_bf16(q, DH, q0, 0, lane);
  const v16bf a_hi = load_a_bf16(q, DH, q0, 32, lane);

  float m_run[8], l_run[8];
#pragma unroll
  for (int j = 0; j < 8; ++j) { m_run[j] = NEG_INF; l_run[j] = 0.0f; }
  v8f acc[4];
#pragma unroll
  for (int t = 0; t < 4; ++t) acc[t] = (v8f){};

  stage_kv(kk, vT, 0, k_sm[0], v_sm[0], tid);         // prime buffer 0

  for (int c = 0; c < S_LEN / 32; ++c) {
    const int c0 = c * 32;
    const int buf = c & 1;
    const bool blk_live = (c0 <= q0max);              // uniform across block

    if (blk_live) {
      wait_async_copies();       // my copies for THIS buffer are done
      __syncthreads();           // everyone's copies done; prev reads done
      if (c0 + 32 <= q0max)      // prefetch next chunk into other buffer
        stage_kv(kk, vT, c0 + 32, k_sm[buf ^ 1], v_sm[buf ^ 1], tid);
    }

    if (c0 > q0 + 15) {
      // Entire chunk above this wave's diagonal: NEG_INF stores only.
      const v4f neg = {NEG_INF, NEG_INF, NEG_INF, NEG_INF};
      float* base = sc + (size_t)(q0 + ll) * S_LEN + c0 + lh * 16;
#pragma unroll
      for (int t = 0; t < 4; ++t) *(v4f*)(base + t * 4) = neg;
      continue;
    }

    // ---- scores: two 16x16 tiles from LDS-staged K ----
    v8f s0 = {}, s1 = {};
    {
      v16bf b0  = load_b_bf16(k_sm[buf], DH, 0, 0, lane);
      v16bf b0h = load_b_bf16(k_sm[buf], DH, 0, 32, lane);
      s0 = wmma_bf16(a_lo, b0, s0);
      s0 = wmma_bf16(a_hi, b0h, s0);
      v16bf b1  = load_b_bf16(k_sm[buf], DH, 16, 0, lane);
      v16bf b1h = load_b_bf16(k_sm[buf], DH, 16, 32, lane);
      s1 = wmma_bf16(a_lo, b1, s1);
      s1 = wmma_bf16(a_hi, b1h, s1);
    }

    // ---- causal mask + score stores (C layout: row=j+8*lh, col=ll) ----
#pragma unroll
    for (int j = 0; j < 8; ++j) {
      int row = q0 + j + lh * 8;
      int col0 = c0 + ll, col1 = c0 + 16 + ll;
      float v0 = (col0 <= row) ? s0[j] : NEG_INF;
      float v1 = (col1 <= row) ? s1[j] : NEG_INF;
      s0[j] = v0; s1[j] = v1;
      sc[(size_t)row * S_LEN + col0] = v0;
      sc[(size_t)row * S_LEN + col1] = v1;
    }

    // ---- online softmax (row stats via 16-lane shfl_xor reductions) ----
    float p0[8], p1[8];
#pragma unroll
    for (int j = 0; j < 8; ++j) {
      float mx = fmaxf(s0[j], s1[j]);
      mx = fmaxf(mx, __shfl_xor(mx, 1));
      mx = fmaxf(mx, __shfl_xor(mx, 2));
      mx = fmaxf(mx, __shfl_xor(mx, 4));
      mx = fmaxf(mx, __shfl_xor(mx, 8));
      float m_new = fmaxf(m_run[j], mx);
      float scale = __expf(m_run[j] - m_new);
      float e0 = __expf(s0[j] - m_new);
      float e1 = __expf(s1[j] - m_new);
      float ss = e0 + e1;
      ss += __shfl_xor(ss, 1);
      ss += __shfl_xor(ss, 2);
      ss += __shfl_xor(ss, 4);
      ss += __shfl_xor(ss, 8);
      l_run[j] = l_run[j] * scale + ss;
      m_run[j] = m_new;
      p0[j] = e0; p1[j] = e1;
#pragma unroll
      for (int t = 0; t < 4; ++t) acc[t][j] *= scale;
    }

    // ---- transpose P (C layout -> A layout) through per-wave LDS ----
    // LDS ops of one wave are in-order (ISA §7); wave_barrier is a free
    // scheduling fence (no block barrier: waves here are divergent).
    __builtin_amdgcn_wave_barrier();
#pragma unroll
    for (int j = 0; j < 8; ++j) {
      pl[(j + lh * 8) * 32 + ll]      = (bf16)p0[j];
      pl[(j + lh * 8) * 32 + 16 + ll] = (bf16)p1[j];
    }
    __builtin_amdgcn_wave_barrier();
    v16bf pa;
    {
      union { v16bf v; v8bf hh[2]; } u;
      const bf16* r = pl + ll * 32 + lh * 8;
      u.hh[0] = *(const v8bf*)(r);
      u.hh[1] = *(const v8bf*)(r + 16);
      pa = u.v;
    }

    // ---- P(16x32) @ V(32x64): four d-tiles from LDS-staged V^T ----
#pragma unroll
    for (int t = 0; t < 4; ++t) {
      v16bf bv_ = load_b_bf16(v_sm[buf], 32, t * 16, 0, lane);
      acc[t] = wmma_bf16(pa, bv_, acc[t]);
    }
  }

  // ---- finalize: divide by l, write merged-head bf16 attn ----
#pragma unroll
  for (int j = 0; j < 8; ++j) {
    float inv = 1.0f / l_run[j];
    int row = b * S_LEN + q0 + j + lh * 8;
#pragma unroll
    for (int t = 0; t < 4; ++t) {
      attn_ws[(size_t)row * DM + h * DH + t * 16 + ll] =
          (bf16)(acc[t][j] * inv);
    }
  }
}

// ---------------------------------------------------------------------------
// Kernel 3: output projection  out = attn @ Wo^T + bo   (fp32 output)
// ---------------------------------------------------------------------------
__global__ __launch_bounds__(256) void out_proj(
    const bf16* __restrict__ attn_ws, const float* __restrict__ Wo,
    const float* __restrict__ bo, float* __restrict__ out) {
  const int lane = threadIdx.x & 31;
  const int wave = threadIdx.x >> 5;
  const int tile = blockIdx.x * 8 + wave;
  const int tn = tile % (DM / 16);
  const int tm = tile / (DM / 16);
  const int r0 = tm * 16, n0 = tn * 16;
  const int ll = lane & 15, lh = lane >> 4;

  v8f acc = {};
  for (int k0 = 0; k0 < DM; k0 += 32) {
    if (k0 + 32 < DM)
      __builtin_prefetch(Wo + (size_t)(n0 + ll) * DM + k0 + 32 + lh * 16, 0, 0);
    v16bf a = load_a_bf16(attn_ws, DM, r0, k0, lane);
    v16bf b = load_b_f32(Wo, DM, n0, k0, lane);
    acc = wmma_bf16(a, b, acc);
  }

  const float bs = bo[n0 + ll];
#pragma unroll
  for (int j = 0; j < 8; ++j) {
    int row = r0 + j + lh * 8;
    out[(size_t)row * DM + n0 + ll] = acc[j] + bs;
  }
}

// ---------------------------------------------------------------------------
extern "C" void kernel_launch(void* const* d_in, const int* in_sizes, int n_in,
                              void* d_out, int out_size, void* d_ws,
                              size_t ws_size, hipStream_t stream) {
  const float* Q  = (const float*)d_in[0];
  const float* K  = (const float*)d_in[1];
  const float* V  = (const float*)d_in[2];
  // d_in[3] = mask (causal, known statically) — ignored
  const float* Wq = (const float*)d_in[4];
  const float* bq = (const float*)d_in[5];
  const float* Wk = (const float*)d_in[6];
  const float* bk = (const float*)d_in[7];
  const float* Wv = (const float*)d_in[8];
  const float* bv = (const float*)d_in[9];
  const float* Wo = (const float*)d_in[10];
  const float* bo = (const float*)d_in[11];

  float* out    = (float*)d_out;                      // [2,2048,768]
  float* scores = out + (size_t)ROWS * DM;            // [2,12,2048,2048]

  bf16* q_ws    = (bf16*)d_ws;                        // [B,H,S,64]
  bf16* k_ws    = q_ws  + (size_t)ROWS * DM;          // [B,H,S,64]
  bf16* vT_ws   = k_ws  + (size_t)ROWS * DM;          // [B,H,64,S]
  bf16* attn_ws = vT_ws + (size_t)ROWS * DM;          // [B*S, 768] merged
  (void)in_sizes; (void)n_in; (void)out_size; (void)ws_size;

  const int tiles = (ROWS / 16) * (DM / 16);          // 12288

  proj_qkv<<<dim3(tiles / 8, 3), 256, 0, stream>>>(
      Q, K, V, Wq, bq, Wk, bk, Wv, bv, q_ws, k_ws, vT_ws);

  attn_kernel<<<dim3(S_LEN / 64, NH, BATCH), 128, 0, stream>>>(
      q_ws, k_ws, vT_ws, scores, attn_ws);

  out_proj<<<tiles / 8, 256, 0, stream>>>(attn_ws, Wo, bo, out);
}
